// TensorBoard_83648783057283
// MI455X (gfx1250) — matrix operand order repaired and checked by reading, not compiled, not run on hardware
//
#include <hip/hip_runtime.h>

// ---------------------------------------------------------------------------
// Go board legality / flood-fill / zobrist kernel for MI455X (gfx1250).
//
// Outputs (concatenated float32 in d_out):
//   [0 .. B*361)         liberties  (WMMA stencil-as-matmul: L = S*E + E*S)
//   [B*361 .. 2*B*361)   legal      (0/1)
//   [2*B*361 .. 3*B*361) groups     (0/1)
//   [3*B*361 .. +B)      new_hash   (int64 cast to float; compile-only env)
//
// d_ws carries the packed 19-bit empty-row masks (B*19 u32 = 1.2 MB) from
// kernel 1 to kernel 2 so the 47 MB stones tensor is read exactly once.
// ---------------------------------------------------------------------------

#define GOH 19
#define GOW 19
#define GOC (GOH * GOW)          // 361
#define MASK19 0x7FFFFu

typedef __attribute__((ext_vector_type(16))) _Float16 v16h;
typedef __attribute__((ext_vector_type(8)))  float    v8f;

// ---------------------------------------------------------------------------
// Kernel 1: one thread per board. Bitboard logic for legal / groups / hash,
// plus export of packed empty rows for the WMMA liberty kernel.
// ---------------------------------------------------------------------------
__global__ __launch_bounds__(128) void go_board_kernel(
    const float* __restrict__ stones,        // [B,2,19,19]
    const int* __restrict__ curp,            // [B]
    const int* __restrict__ kop,             // [B,2]
    const long long* __restrict__ zst,       // [2,19,19]
    const long long* __restrict__ zturn,     // [2]
    const long long* __restrict__ phash,     // [B]
    float* __restrict__ legal_out,           // [B,361]
    float* __restrict__ groups_out,          // [B,361]
    float* __restrict__ hash_out,            // [B]
    unsigned* __restrict__ empws,            // [B,19] packed empty rows (or null)
    int B) {
  // Stage zobrist stone keys in LDS (5776 bytes).
  __shared__ long long zs[2 * GOC];
  for (int i = threadIdx.x; i < 2 * GOC; i += blockDim.x) zs[i] = zst[i];
  __syncthreads();

  int b = blockIdx.x * blockDim.x + threadIdx.x;
  if (b >= B) return;

  const float* s0 = stones + (size_t)b * (2 * GOC);
  const float* s1 = s0 + GOC;
  __builtin_prefetch(s0, 0, 3);   // global_prefetch_b8: pull this board's block

  // --- pack planes into 19-bit row masks, fold zobrist hash on the fly ---
  unsigned pb[GOH], pw[GOH];
  long long h = 0;
#pragma unroll
  for (int r = 0; r < GOH; ++r) {
    unsigned mb = 0, mw = 0;
#pragma unroll
    for (int c = 0; c < GOW; ++c) {
      bool isb = s0[r * GOW + c] > 0.5f;
      bool isw = s1[r * GOW + c] > 0.5f;
      mb |= (unsigned)isb << c;
      mw |= (unsigned)isw << c;
      if (isb) h ^= zs[r * GOW + c];
      if (isw) h ^= zs[GOC + r * GOW + c];
    }
    pb[r] = mb;
    pw[r] = mw;
  }

  unsigned emp[GOH], opp[GOH];
  int cp = curp[b];
#pragma unroll
  for (int r = 0; r < GOH; ++r) {
    emp[r] = ~(pb[r] | pw[r]) & MASK19;
    opp[r] = (cp == 0) ? pw[r] : pb[r];
  }

  // Export packed empty rows for the WMMA liberty kernel (single stones read).
  if (empws) {
    unsigned* wp = empws + (size_t)b * GOH;
#pragma unroll
    for (int r = 0; r < GOH; ++r) wp[r] = emp[r];
  }

  // --- liberties>0 mask and exactly-one-liberty (vulnerable) via bit-sliced add
  unsigned vuln[GOH], legal[GOH];
#pragma unroll
  for (int r = 0; r < GOH; ++r) {
    unsigned up = (r > 0) ? emp[r - 1] : 0u;
    unsigned dn = (r < GOH - 1) ? emp[r + 1] : 0u;
    unsigned lf = (emp[r] << 1) & MASK19;
    unsigned rt = emp[r] >> 1;
    unsigned a = lf ^ rt, ca = lf & rt;
    unsigned bsum = up ^ dn, cb = up & dn;
    unsigned low = a ^ bsum, c2 = a & bsum;
    unsigned exactly1 = low & ~(ca | cb | c2);
    vuln[r]  = opp[r] & exactly1;
    legal[r] = emp[r] & (up | dn | lf | rt);
  }

  // --- capture moves: empty adjacent to a vulnerable opponent stone ---
#pragma unroll
  for (int r = 0; r < GOH; ++r) {
    unsigned dv = ((vuln[r] << 1) & MASK19) | (vuln[r] >> 1)
                | ((r > 0) ? vuln[r - 1] : 0u)
                | ((r < GOH - 1) ? vuln[r + 1] : 0u);
    legal[r] |= emp[r] & dv;
  }

  // --- ko restriction ---
  int kr = kop[2 * b + 0];
  int kc = kop[2 * b + 1];
  if (kr >= 0 && kr < GOH && kc >= 0 && kc < GOW) legal[kr] &= ~(1u << kc);

  // --- flood fill vulnerable seeds within opponent stones (bitmask sweeps) ---
  unsigned g[GOH];
#pragma unroll
  for (int r = 0; r < GOH; ++r) g[r] = vuln[r];
  for (int it = 0; it < 90; ++it) {
    unsigned changed = 0;
#pragma unroll
    for (int r = 0; r < GOH; ++r) {
      unsigned d = ((g[r] << 1) & MASK19) | (g[r] >> 1)
                 | ((r > 0) ? g[r - 1] : 0u)
                 | ((r < GOH - 1) ? g[r + 1] : 0u);
      unsigned e = d & opp[r] & ~g[r];
      g[r] |= e;
      changed |= e;
    }
    if (!changed) break;
  }

  // --- write outputs ---
  float* lo = legal_out + (size_t)b * GOC;
  float* go = groups_out + (size_t)b * GOC;
#pragma unroll
  for (int r = 0; r < GOH; ++r) {
#pragma unroll
    for (int c = 0; c < GOW; ++c) {
      lo[r * GOW + c] = ((legal[r] >> c) & 1u) ? 1.0f : 0.0f;
      go[r * GOW + c] = ((g[r] >> c) & 1u) ? 1.0f : 0.0f;
    }
  }

  // zturn[cp] ^ zturn[1-cp] == zturn[0] ^ zturn[1] for cp in {0,1}
  long long nh = phash[b] ^ h ^ zturn[0] ^ zturn[1];
  hash_out[b] = (float)nh;
}

// ---------------------------------------------------------------------------
// Kernel 2: one wave32 per board. liberties = S*E + E*S with S = 32x32
// tridiagonal shift matrix and E the 32x32 zero-padded empty mask, computed
// with v_wmma_f32_16x16x32_f16 (exact: all values are small integers).
// E arrives as 19 packed u32 rows (from d_ws when available, else rebuilt
// from stones); fragments are bit-extracts, S fragments are hoisted.
// ---------------------------------------------------------------------------
__global__ __launch_bounds__(256) void lib_wmma_kernel(
    const float* __restrict__ stones,      // [B,2,19,19] (fallback path)
    const unsigned* __restrict__ empws,    // [B,19] packed empty rows, may be null
    float* __restrict__ lib_out,           // [B,361]
    int B) {
  __shared__ unsigned Ush[8][32];          // per-wave row words, 1 KB total

  const int lane = threadIdx.x & 31;
  const int wv   = threadIdx.x >> 5;
  const int board = blockIdx.x * 8 + wv;
  if (board >= B) return;  // wave-uniform; active waves keep EXEC all-ones

  // --- lane <-> board row: fetch / build packed empty row ---
  unsigned m = 0;
  if (lane < GOH) {
    if (empws) {
      m = empws[(size_t)board * GOH + lane];   // coalesced within the wave
    } else {
      const float* s0 = stones + (size_t)board * (2 * GOC) + lane * GOW;
      const float* s1 = s0 + GOC;
#pragma unroll
      for (int c = 0; c < GOW; ++c) {
        bool e = (s0[c] < 0.5f) && (s1[c] < 0.5f);
        m |= (unsigned)e << c;
      }
    }
  }
  Ush[wv][lane] = m;  // rows 19..31 are zero (padding)
  // Intra-wave LDS visibility: drain DS counter (cross-lane data flow the
  // compiler cannot see). Lanes execute in lockstep afterwards.
  asm volatile("s_wait_dscnt 0" ::: "memory");

  const int m0 = lane & 15;   // tile row (A/D) / tile col (B/D)
  const int hl = lane >> 4;   // lane half

  // Per-lane gather of the row words this lane's fragment slots touch.
  unsigned wB[16];
#pragma unroll
  for (int idx = 0; idx < 16; ++idx) wB[idx] = Ush[wv][(hl << 4) + idx];
  const unsigned wA[2] = { Ush[wv][m0], Ush[wv][16 + m0] };

  // Hoisted tridiagonal-S fragments: aS depends only on ti, bS only on tj.
  v16h aS[2], bS[2];
#pragma unroll
  for (int t = 0; t < 2; ++t) {
    const int rowA = t * 16 + m0;
    const int colB = t * 16 + m0;
#pragma unroll
    for (int idx = 0; idx < 16; ++idx) {
      // A fragment k mapping (16-bit A 16x32, wave32): K = hl*16 + idx
      const int ka = (hl << 4) + idx;
      // B fragment k mapping (16-bit B 32x16, wave32): K = hl*16 + idx
      const int kb = (hl << 4) + idx;
      aS[t][idx] = (_Float16)(((rowA == ka + 1) || (ka == rowA + 1)) ? 1.0f : 0.0f);
      bS[t][idx] = (_Float16)(((kb == colB + 1) || (colB == kb + 1)) ? 1.0f : 0.0f);
    }
  }

#pragma unroll
  for (int ti = 0; ti < 2; ++ti) {
#pragma unroll
    for (int tj = 0; tj < 2; ++tj) {
      const int colB = tj * 16 + m0;
      v16h aE, bE;
#pragma unroll
      for (int idx = 0; idx < 16; ++idx) {
        const int ka = (hl << 4) + idx;
        aE[idx] = (_Float16)(float)((wA[ti] >> ka) & 1u);
        bE[idx] = (_Float16)(float)((wB[idx] >> colB) & 1u);
      }
      v8f c0 = {};
      // P = S * E  (vertical-neighbor sums)
      v8f p = __builtin_amdgcn_wmma_f32_16x16x32_f16(
          false, aS[ti], false, bE, (short)0, c0, false, false);
      // D = E * S + P  (adds horizontal-neighbor sums)
      v8f d = __builtin_amdgcn_wmma_f32_16x16x32_f16(
          false, aE, false, bS[tj], (short)0, p, false, false);

#pragma unroll
      for (int i = 0; i < 8; ++i) {
        const int row = ti * 16 + (hl << 3) + i;  // D layout: m = i + hl*8
        if (row < GOH && colB < GOW)
          lib_out[(size_t)board * GOC + row * GOW + colB] = d[i];
      }
    }
  }
}

// ---------------------------------------------------------------------------
extern "C" void kernel_launch(void* const* d_in, const int* in_sizes, int n_in,
                              void* d_out, int out_size, void* d_ws, size_t ws_size,
                              hipStream_t stream) {
  (void)n_in; (void)out_size;

  const float*     stones = (const float*)d_in[0];
  const int*       curp   = (const int*)d_in[1];
  const int*       kop    = (const int*)d_in[2];
  const long long* zst    = (const long long*)d_in[3];
  const long long* zturn  = (const long long*)d_in[4];
  const long long* phash  = (const long long*)d_in[5];
  const int B = in_sizes[1];  // current_player has B elements

  float* out = (float*)d_out;
  const size_t plane = (size_t)B * GOC;
  float* lib_out    = out;
  float* legal_out  = out + plane;
  float* groups_out = out + 2 * plane;
  float* hash_out   = out + 3 * plane;

  // Use workspace for the packed empty rows if it is large enough; both
  // branches are deterministic for a given harness configuration.
  unsigned* empws = nullptr;
  if (d_ws && ws_size >= (size_t)B * GOH * sizeof(unsigned))
    empws = (unsigned*)d_ws;

  go_board_kernel<<<(B + 127) / 128, 128, 0, stream>>>(
      stones, curp, kop, zst, zturn, phash,
      legal_out, groups_out, hash_out, empws, B);

  lib_wmma_kernel<<<(B + 7) / 8, 256, 0, stream>>>(stones, empws, lib_out, B);
}
